// NeuralMLPF2_6167573037356
// MI455X (gfx1250) — compile-verified, matrix operand order checked by reading, not run on hardware
//
#include <hip/hip_runtime.h>
#include <hip/hip_bf16.h>
#include <float.h>
#include <limits.h>

// Problem constants (from reference)
#define N_CHAINS 65536
#define BBATCH   64
#define LSEQ     512
#define D_QK     64
#define MAX_SET  8
#define HIDDEN   128
#define IN_DIM   577            // 64 + 8*64 + 1
#define K_PAD    608            // 19 * 32
#define KT_TILES 19
#define CH_PER_BLK 16
#define THREADS  128            // 4 waves of 32 (wave32)

typedef __attribute__((ext_vector_type(16))) _Float16 v16h;
typedef __attribute__((ext_vector_type(8)))  _Float16 v8h;
typedef __attribute__((ext_vector_type(8)))  float    v8f;

union AFrag { v16h v; struct { v8h lo, hi; } s; };

// ---------------------------------------------------------------------------
// Pre-swizzle W1 (577x128 f32, row-major) into per-lane B-fragment order, f16.
// Layout index: ((kt*8 + nt)*32 + lane)*16 + i
// B fragment (16-bit, 32x16 KxN): lane<16 -> N=lane, K in {0..7,16..23};
// lane>=16 -> N=lane-16, K offsets +8.  i = 2*v + h; v<4 -> K=base+2v+h,
// v>=4 -> K=16+base+2(v-4)+h.  K>=577 padded with zero.
// ---------------------------------------------------------------------------
__global__ void prep_w1_kernel(const float* __restrict__ W1,
                               _Float16* __restrict__ W1h) {
    int t = blockIdx.x * blockDim.x + threadIdx.x;
    const int total = KT_TILES * 8 * 32 * 16;
    if (t >= total) return;
    int i    = t & 15;
    int lane = (t >> 4) & 31;
    int nt   = (t >> 9) & 7;
    int kt   = t >> 12;
    int v = i >> 1, h = i & 1;
    int base = (lane < 16) ? 0 : 8;
    int kk = (v < 4) ? (base + 2 * v + h) : (16 + base + 2 * (v - 4) + h);
    int K = kt * 32 + kk;
    int n = nt * 16 + (lane & 15);
    float val = (K < IN_DIM) ? W1[K * HIDDEN + n] : 0.0f;
    W1h[t] = (_Float16)val;
}

// ---------------------------------------------------------------------------
// Main kernel: 16 chains per block, 4 waves.
// ---------------------------------------------------------------------------
__global__ __launch_bounds__(THREADS)
void chain_mlp_kernel(const float* __restrict__ q,
                      const float* __restrict__ k,
                      const int* __restrict__ batch_idx,
                      const unsigned char* __restrict__ mask,
                      const int* __restrict__ count,
                      const float* __restrict__ rank_scores,
                      const _Float16* __restrict__ W1h,
                      const float* __restrict__ b1,
                      const float* __restrict__ W2,
                      const float* __restrict__ b2,
                      float* __restrict__ out) {
    __shared__ float sc[CH_PER_BLK][LSEQ];                       // 32 KB
    __shared__ int   selraw[CH_PER_BLK][MAX_SET];
    __shared__ int   selsort[CH_PER_BLK][MAX_SET];
    __shared__ __align__(16) _Float16 feat[CH_PER_BLK][K_PAD];   // 19 KB
    __shared__ float outacc[CH_PER_BLK];

    const int tid  = threadIdx.x;
    const int lane = tid & 31;
    const int wave = tid >> 5;
    const int c0   = blockIdx.x * CH_PER_BLK;

    // ---- Phase 0: zero feat + outacc, stage masked scores into LDS --------
    for (int idx = tid; idx < CH_PER_BLK * K_PAD; idx += THREADS)
        ((_Float16*)feat)[idx] = (_Float16)0.0f;
    if (tid < CH_PER_BLK) outacc[tid] = 0.0f;

    for (int idx = tid; idx < CH_PER_BLK * LSEQ; idx += THREADS) {
        int m = idx >> 9;           // LSEQ = 512
        int p = idx & (LSEQ - 1);
        int g = (c0 + m) * LSEQ + p;
        sc[m][p] = mask[g] ? rank_scores[g] : -FLT_MAX;
    }
    __syncthreads();

    // ---- Phase 1: 8 rounds of warp-argmax per chain (4 chains per wave) ---
    for (int mm = 0; mm < 4; ++mm) {
        const int m = wave * 4 + mm;
        for (int sel = 0; sel < MAX_SET; ++sel) {
            float best = -FLT_MAX;
            int bidx = LSEQ;
            #pragma unroll
            for (int j = 0; j < 16; ++j) {
                int p = lane + 32 * j;
                float val = sc[m][p];
                if (val > best || (val == best && p < bidx)) { best = val; bidx = p; }
            }
            #pragma unroll
            for (int off = 16; off > 0; off >>= 1) {
                float ov = __shfl_xor(best, off, 32);
                int   oi = __shfl_xor(bidx, off, 32);
                if (ov > best || (ov == best && oi < bidx)) { best = ov; bidx = oi; }
            }
            bool picked = (best > -FLT_MAX);
            if (lane == 0) {
                selraw[m][sel] = picked ? bidx : INT_MAX;
                if (picked) sc[m][bidx] = -FLT_MAX;    // remove winner
            }
            __syncthreads();   // uniform across the block (fixed trip counts)
        }
    }

    // ---- Phase 2: sort picked indices ascending (sentinels to the end) ----
    {
        // one thread per (chain, slot): tid = m*8 + s, exactly 128 threads
        int m = tid >> 3;
        int s = tid & 7;
        int v = selraw[m][s];
        int rank = 0;
        #pragma unroll
        for (int j = 0; j < MAX_SET; ++j) {
            int u = selraw[m][j];
            rank += (u < v) || (u == v && j < s);
        }
        selsort[m][rank] = v;
    }
    __syncthreads();

    // ---- Phase 3: build f16 feature tile [16 x 608] -----------------------
    // q part
    for (int idx = tid; idx < CH_PER_BLK * D_QK; idx += THREADS) {
        int m = idx >> 6, d = idx & 63;
        feat[m][d] = (_Float16)q[(c0 + m) * D_QK + d];
    }
    // packed gathered k
    for (int idx = tid; idx < CH_PER_BLK * MAX_SET * D_QK; idx += THREADS) {
        int m = idx >> 9;
        int r = idx & 511;
        int s = r >> 6, d = r & 63;
        int iv = selsort[m][s];
        float val = 0.0f;
        if (iv < LSEQ) {
            int b = batch_idx[c0 + m];
            val = k[((size_t)b * LSEQ + iv) * D_QK + d];
        }
        feat[m][D_QK + s * D_QK + d] = (_Float16)val;
    }
    // log1p(count)
    if (tid < CH_PER_BLK)
        feat[tid][IN_DIM - 1] = (_Float16)log1pf((float)count[c0 + tid]);
    __syncthreads();

    // ---- Phase 4: feat @ W1 via v_wmma_f32_16x16x32_f16, gelu, @W2 --------
    const int hb   = (lane < 16) ? 0 : 8;   // A-fragment K sub-offset
    const int mrow = lane & 15;             // A-fragment row

    for (int nt2 = 0; nt2 < 2; ++nt2) {
        const int nt = wave * 2 + nt2;               // N-tile 0..7
        const int n  = nt * 16 + (lane & 15);        // this lane's column
        v8f acc = {};
        for (int kt = 0; kt < KT_TILES; ++kt) {
            AFrag a;
            a.s.lo = *(const v8h*)&feat[mrow][kt * 32 + hb];
            a.s.hi = *(const v8h*)&feat[mrow][kt * 32 + 16 + hb];
            v16h bfrag = *(const v16h*)(W1h + (size_t)((kt * 8 + nt) * 32 + lane) * 16);
            acc = __builtin_amdgcn_wmma_f32_16x16x32_f16(
                false, a.v, false, bfrag, (short)0, acc, false, false);
        }
        const float bias = b1[n];
        const float w2v  = W2[n];
        #pragma unroll
        for (int r = 0; r < 8; ++r) {
            float x = acc[r] + bias;
            float g = 0.5f * x * (1.0f + erff(x * 0.70710678118654752f));
            int m = r + ((lane >> 4) << 3);          // C layout: M = r + 8*(lane>=16)
            atomicAdd(&outacc[m], g * w2v);          // ds_add_f32
        }
    }
    __syncthreads();

    // ---- Phase 5: add b2, store -------------------------------------------
    if (tid < CH_PER_BLK)
        out[c0 + tid] = outacc[tid] + b2[0];
}

// ---------------------------------------------------------------------------
extern "C" void kernel_launch(void* const* d_in, const int* in_sizes, int n_in,
                              void* d_out, int out_size, void* d_ws, size_t ws_size,
                              hipStream_t stream) {
    const float*         q           = (const float*)d_in[0];
    const float*         k           = (const float*)d_in[1];
    const int*           batch_idx   = (const int*)d_in[2];
    const unsigned char* mask        = (const unsigned char*)d_in[3];
    const int*           count       = (const int*)d_in[4];
    const float*         rank_scores = (const float*)d_in[5];
    const float*         W1          = (const float*)d_in[6];
    const float*         b1          = (const float*)d_in[7];
    const float*         W2          = (const float*)d_in[8];
    const float*         b2          = (const float*)d_in[9];
    float*               out         = (float*)d_out;

    // Workspace: pre-swizzled W1 in f16 fragment order (19*8*32*16 halves = 152 KB)
    _Float16* W1h = (_Float16*)d_ws;
    const int w1h_elems = KT_TILES * 8 * 32 * 16;
    prep_w1_kernel<<<(w1h_elems + 255) / 256, 256, 0, stream>>>(W1, W1h);

    const int nblocks = N_CHAINS / CH_PER_BLK;   // 4096
    chain_mlp_kernel<<<nblocks, THREADS, 0, stream>>>(
        q, k, batch_idx, mask, count, rank_scores, W1h, b1, W2, b2, out);
}
